// FeatureAdaption_5720896438909
// MI455X (gfx1250) — compile-verified, matrix operand order checked by reading, not run on hardware
//
#include <hip/hip_runtime.h>
#include <hip/hip_bf16.h>

// Problem constants
#define B_  8
#define G_  3
#define IN_ 384
#define OUT_ 384
#define H_  64
#define W_  64
#define CG_ 128          // IN/G
#define OG_ 128          // OUT/G
#define KDIM 1152        // CG*9
#define KT_  36          // KDIM/32 k-tiles total
#define KCH  18          // k-tiles per staged chunk (2 chunks)
#define MT_  8           // OG/16 m-tiles
#define HW_  4096        // H*W
#define NPIX 32          // pixels per workgroup tile (2 N-tiles of 16)
#define NPAIR (NPIX * 9) // 288 (pixel, tap) pairs
#define HALF_STRIDE (KCH * 32 * 16)       // ushorts per 16-pixel half per chunk
#define CHUNK_ELEMS (KCH * 32 * NPIX)     // 18432 staged elements per chunk

typedef __attribute__((ext_vector_type(16))) __bf16 v16bf;
typedef __attribute__((ext_vector_type(8)))  float  v8f;

__device__ __forceinline__ unsigned short f2bf(float f) {
    unsigned u = __float_as_uint(f);
    unsigned r = u + 0x7FFFu + ((u >> 16) & 1u);   // round-to-nearest-even
    return (unsigned short)(r >> 16);
}

__device__ __forceinline__ float corner(const float* __restrict__ pl, int y, int x) {
    bool valid = (y >= 0) && (y < H_) && (x >= 0) && (x < W_);
    int yc = min(max(y, 0), H_ - 1);
    int xc = min(max(x, 0), W_ - 1);
    float v = pl[yc * W_ + xc];
    return valid ? v : 0.0f;
}

// ---------------------------------------------------------------------------
// Prep: convert w_d (fp32, [OUT][CG][3][3]) -> bf16 in WMMA A-fragment order:
//   wfrag[(((g*8 + mt)*36 + kt)*32 + lane)*16 + h]
// (lane,h) -> (m = lane&15, k_local) per CDNA5 16-bit A 16x32 layout:
//   k_local = (h&7) | ((h>>3)<<4) | (((lane>>4)&1)<<3)
// ---------------------------------------------------------------------------
__global__ __launch_bounds__(256)
void fa_prep_weights(const float* __restrict__ w_d, unsigned short* __restrict__ wfrag) {
    int idx = blockIdx.x * 256 + threadIdx.x;
    if (idx >= G_ * MT_ * KT_ * 32 * 16) return;
    int hh   = idx & 15;
    int lane = (idx >> 4) & 31;
    int tile = idx >> 9;            // (g*8+mt)*36 + kt
    int kt   = tile % KT_;
    int gm   = tile / KT_;          // g*8 + mt
    int g    = gm >> 3;
    int mt   = gm & 7;
    int kl   = (hh & 7) | ((hh >> 3) << 4) | (((lane >> 4) & 1) << 3);
    int k    = kt * 32 + kl;                       // 0..1151  (= c*9 + tap)
    int o    = g * OG_ + mt * 16 + (lane & 15);    // global output channel
    wfrag[idx] = f2bf(w_d[(size_t)o * KDIM + k]);
}

// ---------------------------------------------------------------------------
// Main fused kernel. Grid: (HW/32, G, B), Block: 256 (8 waves).
// Each wave owns one 16-channel M-tile and TWO 16-pixel N-tiles (one A-frag
// load feeds two WMMAs). K is processed in two staged chunks of 576 so the
// LDS footprint stays at 41.5 KB (static, safe under any per-WG limit).
// ---------------------------------------------------------------------------
__global__ __launch_bounds__(256)
void fa_main(const float* __restrict__ input,
             const float* __restrict__ wh_pred,
             const float* __restrict__ w_off,
             const float* __restrict__ b_off,
             const float* __restrict__ b_d,
             const unsigned short* __restrict__ wfrag,
             float* __restrict__ out) {
    __shared__ __align__(16) unsigned short sS[2 * HALF_STRIDE];  // 36.9 KB
    __shared__ float sWy[NPAIR], sWx[NPAIR];
    __shared__ int   sY0[NPAIR], sX0[NPAIR];

    const int tix = blockIdx.x;         // pixel tile (0..127)
    const int g   = blockIdx.y;
    const int b   = blockIdx.z;
    const int p0  = tix * NPIX;
    const int h   = p0 >> 6;            // / W_
    const int w0  = p0 & (W_ - 1);      // 0 or 32
    const int tid = threadIdx.x;

    // ---- Stage 1: per-(pixel, tap) sampling coordinates ----
    for (int i = tid; i < NPAIR; i += 256) {
        int n = i / 9;
        int t = i - n * 9;
        int wpix = w0 + n;
        const float* whp = wh_pred + ((size_t)(b * 2 * G_ + 2 * g)) * HW_ + h * W_ + wpix;
        float wh0 = whp[0];
        float wh1 = whp[HW_];
        const float* wo = w_off + g * 36;   // (18,2)
        const float* bo = b_off + g * 18;
        float offy = wh0 * wo[4 * t + 0] + wh1 * wo[4 * t + 1] + bo[2 * t + 0];
        float offx = wh0 * wo[4 * t + 2] + wh1 * wo[4 * t + 3] + bo[2 * t + 1];
        float py = offy + (float)h    + (float)(t / 3 - 1);
        float px = offx + (float)wpix + (float)(t % 3 - 1);
        float y0 = floorf(py);
        float x0 = floorf(px);
        sWy[i] = py - y0;
        sWx[i] = px - x0;
        sY0[i] = (int)y0;
        sX0[i] = (int)x0;
    }
    __syncthreads();

    const float* gbase = input + ((size_t)(b * IN_ + g * CG_)) * HW_;
    const int wave = tid >> 5;
    const int lane = tid & 31;
    v8f acc0 = {};
    v8f acc1 = {};
    const unsigned short* wbase =
        wfrag + ((size_t)(g * MT_ + wave) * KT_ * 32 + lane) * 16;
    const unsigned short* sb0 = &sS[lane * 16];
    const unsigned short* sb1 = &sS[HALF_STRIDE + lane * 16];

    for (int ch = 0; ch < 2; ++ch) {
        // ---- Stage 2: bilinear sample this K-chunk into LDS (B-frag layout) ----
        for (int i = tid; i < CHUNK_ELEMS; i += 256) {
            int n   = i & 31;               // pixel within tile
            int klc = i >> 5;               // 0..575 within chunk
            int k   = ch * (KCH * 32) + klc;
            int c = k / 9;
            int t = k - c * 9;
            int pr = n * 9 + t;
            int y0 = sY0[pr], x0 = sX0[pr];
            float wy = sWy[pr], wx = sWx[pr];
            const float* pl = gbase + (size_t)c * HW_;
            float v00 = corner(pl, y0,     x0);
            float v01 = corner(pl, y0,     x0 + 1);
            float v10 = corner(pl, y0 + 1, x0);
            float v11 = corner(pl, y0 + 1, x0 + 1);
            float v = v00 * (1.0f - wy) * (1.0f - wx)
                    + v01 * (1.0f - wy) * wx
                    + v10 * wy * (1.0f - wx)
                    + v11 * wy * wx;
            // B-fragment swizzle within each 16-pixel half:
            //   lane = (n&15) + 16*bit3(k_local), h = (k&7)|((k&16)>>1)
            int half = n >> 4;
            int kll  = klc & 31;
            int kt   = klc >> 5;            // 0..17
            int slane = (n & 15) + (((kll >> 3) & 1) << 4);
            int hh    = (kll & 7) | ((kll & 16) >> 1);
            sS[half * HALF_STRIDE + (kt * 32 + slane) * 16 + hh] = f2bf(v);
        }
        __syncthreads();

        // ---- Stage 3: WMMA over this chunk. One A-frag load -> two WMMAs. ----
        const unsigned short* wch = wbase + (size_t)ch * KCH * 512;
#pragma unroll 2
        for (int kt = 0; kt < KCH; ++kt) {
            v16bf a   = *(const v16bf*)(wch + (size_t)kt * 512);   // 2x global_load_b128
            v16bf bb0 = *(const v16bf*)(sb0 + (size_t)kt * 512);   // 2x ds_load_b128
            v16bf bb1 = *(const v16bf*)(sb1 + (size_t)kt * 512);   // 2x ds_load_b128
            acc0 = __builtin_amdgcn_wmma_f32_16x16x32_bf16(
                false, a, false, bb0, (short)0, acc0, false, false);
            acc1 = __builtin_amdgcn_wmma_f32_16x16x32_bf16(
                false, a, false, bb1, (short)0, acc1, false, false);
        }
        __syncthreads();   // protect sS before restaging next chunk
    }

    // ---- Epilogue: bias + store. D layout: lanes 0-15 -> M=r, lanes 16-31 -> M=8+r
    int ncol  = lane & 15;
    int mbase = wave * 16 + ((lane >> 4) << 3);
    int ch0   = g * OG_ + mbase;
    float* op0 = out + ((size_t)b * OUT_ + ch0) * HW_ + h * W_ + (w0 + ncol);
    float* op1 = op0 + 16;
#pragma unroll
    for (int r = 0; r < 8; ++r) {
        float bias = b_d[ch0 + r];
        op0[(size_t)r * HW_] = acc0[r] + bias;
        op1[(size_t)r * HW_] = acc1[r] + bias;
    }
}

extern "C" void kernel_launch(void* const* d_in, const int* in_sizes, int n_in,
                              void* d_out, int out_size, void* d_ws, size_t ws_size,
                              hipStream_t stream) {
    (void)in_sizes; (void)n_in; (void)out_size; (void)ws_size;
    const float* input   = (const float*)d_in[0];
    const float* wh_pred = (const float*)d_in[1];
    const float* w_off   = (const float*)d_in[2];
    const float* b_off   = (const float*)d_in[3];
    const float* w_d     = (const float*)d_in[4];
    const float* b_d     = (const float*)d_in[5];
    float* out = (float*)d_out;
    unsigned short* wfrag = (unsigned short*)d_ws;   // needs 884,736 bytes

    const int welems = G_ * MT_ * KT_ * 32 * 16;     // 442368
    fa_prep_weights<<<(welems + 255) / 256, 256, 0, stream>>>(w_d, wfrag);

    dim3 grid(HW_ / NPIX, G_, B_);                   // (128, 3, 8)
    fa_main<<<grid, 256, 0, stream>>>(input, wh_pred, w_off, b_off, b_d, wfrag, out);
}
